// GlmImageVisionModel_34694745817247
// MI455X (gfx1250) — matrix operand order repaired and checked by reading, not compiled, not run on hardware
//
#include <hip/hip_runtime.h>
#include <hip/hip_bf16.h>
#include <math.h>

// ---------------------------------------------------------------------------
// GLM vision tower forward for MI455X (gfx1250, wave32, WMMA bf16).
// All GEMMs (patch-embed, QKV, attention QK^T / PV, proj, FC1, FC2) run on
// v_wmma_f32_16x16x32_bf16 with f32 accumulation.  Weights are converted
// fp32->bf16 once per call into the workspace (96MB bf16, L2-resident).
// GEMM: 64x256 block tile, double-buffered LDS, one barrier per 32-K slab.
// ---------------------------------------------------------------------------

typedef __bf16 bf16;
typedef __attribute__((ext_vector_type(16))) __bf16 v16bf;
typedef __attribute__((ext_vector_type(8)))  float  v8f;

#define D_     1024
#define NH_    16
#define HD_    64
#define FF_    4096
#define DEPTH_ 4
#define S_     4096
#define L_     1024
#define KIN_   588
#define KINP_  608          // 588 padded to a multiple of 32
#define SCALE_ 0.125f       // HD^-0.5

__device__ __forceinline__ v8f vzero8() {
  v8f z;
#pragma unroll
  for (int i = 0; i < 8; ++i) z[i] = 0.0f;
  return z;
}

__device__ __forceinline__ v8f wmma_bf16(v16bf a, v16bf b, v8f c) {
  // (neg_a, A, neg_b, B, c_mod, C, reuse_a, reuse_b)
  return __builtin_amdgcn_wmma_f32_16x16x32_bf16(false, a, false, b, (short)0, c,
                                                 false, false);
}

// ---------------------------------------------------------------------------
// fp32 -> bf16 conversion kernels
// ---------------------------------------------------------------------------
__global__ void cvt_bf16_kernel(const float* __restrict__ in, bf16* __restrict__ out,
                                long long n4) {  // n4 = n/4, n % 4 == 0
  long long i = (long long)blockIdx.x * blockDim.x + threadIdx.x;
  const long long stride = (long long)gridDim.x * blockDim.x;
  for (; i < n4; i += stride) {
    const float4 v = ((const float4*)in)[i];
    union { bf16 h[4]; uint2 u; } p;
    p.h[0] = (bf16)v.x; p.h[1] = (bf16)v.y; p.h[2] = (bf16)v.z; p.h[3] = (bf16)v.w;
    ((uint2*)out)[i] = p.u;
  }
}

__global__ void cvt_pad_kernel(const float* __restrict__ in, bf16* __restrict__ out,
                               int rows, int K, int KP) {
  const long long n = (long long)rows * KP;
  long long i = (long long)blockIdx.x * blockDim.x + threadIdx.x;
  const long long stride = (long long)gridDim.x * blockDim.x;
  for (; i < n; i += stride) {
    const long long r = i / KP;
    const int k = (int)(i - r * KP);
    out[i] = (k < K) ? (bf16)in[r * K + k] : (bf16)0.0f;
  }
}

// ---------------------------------------------------------------------------
// Generic bf16 WMMA GEMM:  C[M,N] = A[M,K] * W[N,K]^T  (+bias, gelu, residual)
// Block = 256 threads (8 waves); block tile M=64 x N=256; wave tile 32x64
// (2x4 WMMA accumulators).  Double-buffered LDS staging in *fragment order*
// (per 16-wide tile: 32 lanes x 16 bf16, one contiguous 32B chunk per lane)
// gives exactly one barrier per 32-K slab, with the global loads for slab
// k+1 issued before the WMMAs of slab k and the ds_stores after them.
// ---------------------------------------------------------------------------
template <bool GELU, bool HAS_RES, bool HAS_OUTF, bool HAS_OUTB>
__global__ __launch_bounds__(256) void gemm_bf16_kernel(
    const bf16* __restrict__ A, const bf16* __restrict__ W,
    const float* __restrict__ bias, const float* __restrict__ resid,
    float* __restrict__ outf, bf16* __restrict__ outb,
    int M, int N, int K) {
  __shared__ __align__(32) bf16 alds[2][4 * 32 * 16];    // 2 x 4 M-tiles
  __shared__ __align__(32) bf16 blds[2][16 * 32 * 16];   // 2 x 16 N-tiles

  const int tid  = threadIdx.x;
  const int lane = tid & 31;
  const int wid  = tid >> 5;
  const int mw   = wid >> 2;          // 0..1  (wave M group: mtiles mw*2,mw*2+1)
  const int nw   = wid & 3;           // 0..3  (wave N group: ntiles nw*4..nw*4+3)
  const int m0   = blockIdx.y * 64;
  const int n0   = blockIdx.x * 256;

  v8f acc[2][4];
#pragma unroll
  for (int mi = 0; mi < 2; ++mi)
#pragma unroll
    for (int ni = 0; ni < 4; ++ni) acc[mi][ni] = vzero8();

  // A staging: thread -> (mtile, lane, half) ; two 16B chunks form one frag.
  const int amt = tid >> 6;           // 0..3
  const int ar  = tid & 63;
  const int aln = ar >> 1;            // fragment lane 0..31
  const int aj  = ar & 1;             // which 8-elem half
  const bf16* aptr = A + (size_t)(m0 + amt * 16 + (aln & 15)) * K +
                     aj * 8 + (aln >> 4) * 16;
  const int aslot = (amt * 32 + aln) * 16 + aj * 8;

  // B staging: thread -> (ntiles wid and wid+8, lane); one 32B chunk each.
  const bf16* bptr0 = W + (size_t)(n0 + wid * 16 + (lane & 15)) * K +
                      (lane >> 4) * 16;
  const bf16* bptr1 = bptr0 + (size_t)128 * K;   // +8 ntiles = +128 columns
  const int bslot0 = (wid * 32 + lane) * 16;
  const int bslot1 = ((wid + 8) * 32 + lane) * 16;

  // prime buffer 0
  uint4 areg  = *(const uint4*)aptr;
  v16bf breg0 = *(const v16bf*)bptr0;
  v16bf breg1 = *(const v16bf*)bptr1;
  *(uint4*)&alds[0][aslot]  = areg;
  *(v16bf*)&blds[0][bslot0] = breg0;
  *(v16bf*)&blds[0][bslot1] = breg1;

  int cur = 0;
  for (int k0 = 0; k0 < K; k0 += 32) {
    __syncthreads();                        // buf[cur] visible; buf[cur^1] free
    const bool hasNext = (k0 + 32 < K);
    if (hasNext) {                          // issue next-slab global loads now
      areg  = *(const uint4*)(aptr + k0 + 32);
      breg0 = *(const v16bf*)(bptr0 + k0 + 32);
      breg1 = *(const v16bf*)(bptr1 + k0 + 32);
      __builtin_prefetch(bptr0 + k0 + 160, 0, 1);  // global_prefetch_b8
      __builtin_prefetch(bptr1 + k0 + 160, 0, 1);
    }

    v16bf afr[2], bfr[4];
#pragma unroll
    for (int mi = 0; mi < 2; ++mi)
      afr[mi] = *(const v16bf*)&alds[cur][((mw * 2 + mi) * 32 + lane) * 16];
#pragma unroll
    for (int ni = 0; ni < 4; ++ni)
      bfr[ni] = *(const v16bf*)&blds[cur][((nw * 4 + ni) * 32 + lane) * 16];
#pragma unroll
    for (int mi = 0; mi < 2; ++mi)
#pragma unroll
      for (int ni = 0; ni < 4; ++ni)
        acc[mi][ni] = wmma_bf16(afr[mi], bfr[ni], acc[mi][ni]);

    if (hasNext) {                          // stash next slab into other buffer
      const int nxt = cur ^ 1;
      *(uint4*)&alds[nxt][aslot]  = areg;
      *(v16bf*)&blds[nxt][bslot0] = breg0;
      *(v16bf*)&blds[nxt][bslot1] = breg1;
    }
    cur ^= 1;
  }

  // Epilogue: C layout -> row = r + 8*(lane>=16), col = lane%16. Branch-free.
#pragma unroll
  for (int mi = 0; mi < 2; ++mi)
#pragma unroll
    for (int ni = 0; ni < 4; ++ni) {
      const int row0 = m0 + (mw * 2 + mi) * 16 + ((lane >> 4) << 3);
      const int col  = n0 + (nw * 4 + ni) * 16 + (lane & 15);
      const float bv = bias[col];
#pragma unroll
      for (int r = 0; r < 8; ++r) {
        float v = acc[mi][ni][r] + bv;
        if constexpr (GELU) v = 0.5f * v * (1.0f + erff(v * 0.70710678118f));
        const size_t idx = (size_t)(row0 + r) * N + col;
        if constexpr (HAS_RES)  v += resid[idx];
        if constexpr (HAS_OUTF) outf[idx] = v;
        if constexpr (HAS_OUTB) outb[idx] = (bf16)v;
      }
    }
}

// ---------------------------------------------------------------------------
// LayerNorm (fp32 in, bf16 out), one block per row of 1024.  Wave32 shfl_xor
// butterflies reduce sum(x) and sum(x^2) together; one LDS combine.
// ---------------------------------------------------------------------------
__global__ __launch_bounds__(256) void layernorm_kernel(
    const float* __restrict__ x, const float* __restrict__ g,
    const float* __restrict__ b, bf16* __restrict__ out) {
  __shared__ float wsum[8], wsq[8];
  const int row = blockIdx.x, tid = threadIdx.x;
  const int lane = tid & 31, wid = tid >> 5;
  const float* xr = x + (size_t)row * D_;
  float v[4];
  float s = 0.0f, s2 = 0.0f;
#pragma unroll
  for (int i = 0; i < 4; ++i) {
    v[i] = xr[tid + i * 256];
    s += v[i];
    s2 += v[i] * v[i];
  }
#pragma unroll
  for (int o = 16; o > 0; o >>= 1) {
    s  += __shfl_xor(s, o, 32);
    s2 += __shfl_xor(s2, o, 32);
  }
  if (lane == 0) { wsum[wid] = s; wsq[wid] = s2; }
  __syncthreads();
  float ts = 0.0f, ts2 = 0.0f;
#pragma unroll
  for (int i = 0; i < 8; ++i) { ts += wsum[i]; ts2 += wsq[i]; }
  const float mean = ts * (1.0f / D_);
  const float var  = ts2 * (1.0f / D_) - mean * mean;
  const float rstd = rsqrtf(var + 1e-6f);
  bf16* orow = out + (size_t)row * D_;
#pragma unroll
  for (int i = 0; i < 4; ++i) {
    const int c = tid + i * 256;
    orow[c] = (bf16)((v[i] - mean) * rstd * g[c] + b[c]);
  }
}

// ---------------------------------------------------------------------------
// Bilinear (border-pad, align_corners=False) 2D pos-embed add, one block/token.
// ---------------------------------------------------------------------------
__global__ __launch_bounds__(256) void add_pos_kernel(float* __restrict__ x,
                                                      const float* __restrict__ pe) {
  const int s = blockIdx.x;
  const int l = s & (L_ - 1);
  const int g0 = l >> 6, g2 = (l >> 2) & 15, g1 = (l >> 1) & 1, g3 = l & 1;
  const int hh = g0 * 2 + g1;   // hpos in [0,32)
  const int ww = g2 * 2 + g3;   // wpos in [0,32)
  // (n+1)*0.5*24 - 0.5 with n = (p+0.5)/32*2-1  ->  (p+0.5)*0.75 - 0.5
  const float fx = (ww + 0.5f) * 0.75f - 0.5f;
  const float fy = (hh + 0.5f) * 0.75f - 0.5f;
  const int x0 = (int)floorf(fx), y0 = (int)floorf(fy);
  const float wx = fx - (float)x0, wy = fy - (float)y0;
  const int x0i = min(max(x0, 0), 23), x1i = min(max(x0 + 1, 0), 23);
  const int y0i = min(max(y0, 0), 23), y1i = min(max(y0 + 1, 0), 23);
  const float* p00 = pe + (size_t)(y0i * 24 + x0i) * D_;
  const float* p01 = pe + (size_t)(y0i * 24 + x1i) * D_;
  const float* p10 = pe + (size_t)(y1i * 24 + x0i) * D_;
  const float* p11 = pe + (size_t)(y1i * 24 + x1i) * D_;
  const float w00 = (1.0f - wy) * (1.0f - wx), w01 = (1.0f - wy) * wx;
  const float w10 = wy * (1.0f - wx), w11 = wy * wx;
  float* xr = x + (size_t)s * D_;
  for (int c = threadIdx.x; c < D_; c += 256)
    xr[c] += p00[c] * w00 + p01[c] * w01 + p10[c] * w10 + p11[c] * w11;
}

// ---------------------------------------------------------------------------
// Attention: one block per (q-tile of 32 rows, head, image). 128 threads.
// Full 32x1024 f32 score slab lives in LDS, softmax in place, P re-emitted as
// bf16 WMMA A-fragments.  PV: each wave owns one 16-wide output column group
// (no cross-wave reduction); V is staged with coalesced 128B row loads and
// scattered once into B-fragment order in LDS (reusing the dead score slab).
// ---------------------------------------------------------------------------
#define ATTN_SMEM (131072 + 65536 + 4096)

__global__ __launch_bounds__(128) void attention_kernel(const bf16* __restrict__ qkv,
                                                        bf16* __restrict__ obuf) {
  extern __shared__ char smem[];
  float* sc    = (float*)smem;                       // [32][1024] scores
  bf16*  pfrag = (bf16*)(smem + 131072);             // [2][32][32][16] P frags
  bf16*  qfrag = (bf16*)(smem + 131072 + 65536);     // [2][2][32][16] Q frags
  bf16*  vfrag = (bf16*)smem;                        // reuse: [4][4][32][16]
  __shared__ float redsm[32][4];

  const int tid  = threadIdx.x;
  const int lane = tid & 31;
  const int w    = tid >> 5;
  const int qt = blockIdx.x, h = blockIdx.y, bimg = blockIdx.z;
  const int rowbase = bimg * L_ + qt * 32;
  const int RS = 3 * D_;  // qkv row stride

  // ---- stage Q fragments (2 mtiles x 2 ktiles) ----
  {
    const int mt = tid >> 6;
    const int r2 = tid & 63;
    const int ln = r2 >> 1, j = r2 & 1;
    const bf16* src = qkv + (size_t)(rowbase + mt * 16 + (ln & 15)) * RS + h * HD_;
#pragma unroll
    for (int kt = 0; kt < 2; ++kt) {
      const int koff = kt * 32 + j * 8 + (ln >> 4) * 16;
      *(uint4*)&qfrag[((mt * 2 + kt) * 32 + ln) * 16 + j * 8] =
          *(const uint4*)&src[koff];
    }
  }
  __syncthreads();

  v16bf qa[2][2];
#pragma unroll
  for (int mt = 0; mt < 2; ++mt)
#pragma unroll
    for (int kt = 0; kt < 2; ++kt)
      qa[mt][kt] = *(const v16bf*)&qfrag[((mt * 2 + kt) * 32 + lane) * 16];

  // ---- scores = Q K^T * scale ----
  const bf16* kbase = qkv + (size_t)(bimg * L_) * RS + D_ + h * HD_;
  for (int nt = w; nt < 64; nt += 4) {
    v16bf bk[2];
#pragma unroll
    for (int kt = 0; kt < 2; ++kt)
      bk[kt] = *(const v16bf*)&kbase[(size_t)(nt * 16 + (lane & 15)) * RS +
                                     kt * 32 + (lane >> 4) * 16];
#pragma unroll
    for (int mt = 0; mt < 2; ++mt) {
      v8f acc = vzero8();
      acc = wmma_bf16(qa[mt][0], bk[0], acc);
      acc = wmma_bf16(qa[mt][1], bk[1], acc);
      const int r0  = mt * 16 + ((lane >> 4) << 3);
      const int col = nt * 16 + (lane & 15);
#pragma unroll
      for (int r = 0; r < 8; ++r) sc[(r0 + r) * L_ + col] = acc[r] * SCALE_;
    }
  }
  __syncthreads();

  // ---- softmax over 1024, emit bf16 P fragments ----
  {
    const int row = tid >> 2, seg = tid & 3;
    float* srow = sc + row * L_;
    float mx = -3.0e38f;
    for (int c = seg * 256; c < seg * 256 + 256; ++c) mx = fmaxf(mx, srow[c]);
    redsm[row][seg] = mx;
    __syncthreads();
    mx = fmaxf(fmaxf(redsm[row][0], redsm[row][1]),
               fmaxf(redsm[row][2], redsm[row][3]));
    float sum = 0.0f;
    for (int c = seg * 256; c < seg * 256 + 256; ++c) {
      const float e = __expf(srow[c] - mx);
      srow[c] = e;
      sum += e;
    }
    __syncthreads();
    redsm[row][seg] = sum;
    __syncthreads();
    sum = redsm[row][0] + redsm[row][1] + redsm[row][2] + redsm[row][3];
    const float inv = 1.0f / sum;
    const int mt = row >> 4, m = row & 15;
    for (int c = seg * 256; c < seg * 256 + 256; ++c) {
      const int kt = c >> 5, kk = c & 31;
      const int g = kk >> 4;
      const int j = kk & 15;
      pfrag[((mt * 32 + kt) * 32 + (m + (g << 4))) * 16 + j] = (bf16)(srow[c] * inv);
    }
  }
  __syncthreads();

  // ---- O = P V : wave w owns output columns [w*16, w*16+16) ----
  const bf16* vbase = qkv + (size_t)(bimg * L_) * RS + 2 * D_ + h * HD_;
  v8f oacc[2];
  oacc[0] = vzero8();
  oacc[1] = vzero8();
  // staging decomposition: this thread moves one V row per 128-row group
  const int sktL = tid >> 5;         // local ktile 0..3
  const int skk  = tid & 31;         // k within ktile
  const int sg   = skk >> 4;         // lane-group select
  const int sj   = skk & 15;         // fragment element index

  for (int ktg = 0; ktg < 8; ++ktg) {
    __syncthreads();  // previous vfrag reads complete
    {
      const int k = ktg * 128 + tid;                 // global k row (0..1023)
      const bf16* vrow = vbase + (size_t)k * RS;     // 64 contiguous bf16
      uint4 buf[8];
#pragma unroll
      for (int u = 0; u < 8; ++u) buf[u] = ((const uint4*)vrow)[u];
      const bf16* be = (const bf16*)buf;
#pragma unroll
      for (int d = 0; d < 64; ++d) {
        const int nt    = d >> 4;
        const int lanei = (d & 15) | (sg << 4);
        vfrag[((sktL * 4 + nt) * 32 + lanei) * 16 + sj] = be[d];
      }
    }
    __syncthreads();  // vfrag staged
#pragma unroll
    for (int ktL = 0; ktL < 4; ++ktL) {
      const int kt = ktg * 4 + ktL;
      const v16bf pa0 = *(const v16bf*)&pfrag[((0 * 32 + kt) * 32 + lane) * 16];
      const v16bf pa1 = *(const v16bf*)&pfrag[((1 * 32 + kt) * 32 + lane) * 16];
      const v16bf bv  = *(const v16bf*)&vfrag[((ktL * 4 + w) * 32 + lane) * 16];
      oacc[0] = wmma_bf16(pa0, bv, oacc[0]);
      oacc[1] = wmma_bf16(pa1, bv, oacc[1]);
    }
  }

  // write O (bf16) straight into proj's A-activation buffer
#pragma unroll
  for (int mt = 0; mt < 2; ++mt) {
    const int r0  = mt * 16 + ((lane >> 4) << 3);
    const int col = h * HD_ + w * 16 + (lane & 15);
#pragma unroll
    for (int r = 0; r < 8; ++r)
      obuf[(size_t)(rowbase + r0 + r) * D_ + col] = (bf16)oacc[mt][r];
  }
}

// ---------------------------------------------------------------------------
// Host-side orchestration
// ---------------------------------------------------------------------------
extern "C" void kernel_launch(void* const* d_in, const int* in_sizes, int n_in,
                              void* d_out, int out_size, void* d_ws, size_t ws_size,
                              hipStream_t stream) {
  (void)in_sizes; (void)n_in; (void)out_size; (void)ws_size;
  const float* pixel   = (const float*)d_in[0];
  // d_in[1] = grid_thw (static metadata, unused)
  const float* conv_w  = (const float*)d_in[2];
  const float* conv_b  = (const float*)d_in[3];
  const float* pos_emb = (const float*)d_in[4];
  const float* ln1_g   = (const float*)d_in[5];
  const float* ln1_b   = (const float*)d_in[6];
  const float* qkv_w   = (const float*)d_in[7];
  const float* qkv_b   = (const float*)d_in[8];
  const float* proj_w  = (const float*)d_in[9];
  const float* proj_b  = (const float*)d_in[10];
  const float* ln2_g   = (const float*)d_in[11];
  const float* ln2_b   = (const float*)d_in[12];
  const float* fc1_w   = (const float*)d_in[13];
  const float* fc1_b   = (const float*)d_in[14];
  const float* fc2_w   = (const float*)d_in[15];
  const float* fc2_b   = (const float*)d_in[16];

  char* ws = (char*)d_ws;
  size_t off = 0;
  auto take = [&](size_t bytes) -> char* {
    char* p = ws + off;
    off = (off + bytes + 255) & ~(size_t)255;
    return p;
  };

  bf16* wqkv  = (bf16*)take(sizeof(bf16) * 4ull * 3 * D_ * D_);
  bf16* wproj = (bf16*)take(sizeof(bf16) * 4ull * D_ * D_);
  bf16* wfc1  = (bf16*)take(sizeof(bf16) * 4ull * FF_ * D_);
  bf16* wfc2  = (bf16*)take(sizeof(bf16) * 4ull * D_ * FF_);
  bf16* wconv = (bf16*)take(sizeof(bf16) * (size_t)D_ * KINP_);
  bf16* pixb  = (bf16*)take(sizeof(bf16) * (size_t)S_ * KINP_);
  float* xf   = (float*)take(sizeof(float) * (size_t)S_ * D_);
  bf16* lnb   = (bf16*)take(sizeof(bf16) * (size_t)S_ * D_);
  bf16* qkvb  = (bf16*)take(sizeof(bf16) * (size_t)S_ * 3 * D_);
  bf16* ob    = (bf16*)take(sizeof(bf16) * (size_t)S_ * D_);
  bf16* hb    = (bf16*)take(sizeof(bf16) * (size_t)S_ * FF_);

  // Weight conversion fp32 -> bf16 (each call; deterministic).
  cvt_bf16_kernel<<<2048, 256, 0, stream>>>(qkv_w, wqkv, 3ll * D_ * D_);       // 4*3M/4
  cvt_bf16_kernel<<<2048, 256, 0, stream>>>(proj_w, wproj, 1ll * D_ * D_);     // 4*1M/4
  cvt_bf16_kernel<<<2048, 256, 0, stream>>>(fc1_w, wfc1, 4ll * FF_ * D_ / 4);
  cvt_bf16_kernel<<<2048, 256, 0, stream>>>(fc2_w, wfc2, 4ll * D_ * FF_ / 4);
  cvt_pad_kernel<<<1024, 256, 0, stream>>>(conv_w, wconv, D_, KIN_, KINP_);
  cvt_pad_kernel<<<2048, 256, 0, stream>>>(pixel, pixb, S_, KIN_, KINP_);

  // Patch embed (GEMM, K padded to 608) + interpolated position embedding.
  gemm_bf16_kernel<false, false, true, false>
      <<<dim3(D_ / 256, S_ / 64), 256, 0, stream>>>(
          pixb, wconv, conv_b, nullptr, xf, nullptr, S_, D_, KINP_);
  add_pos_kernel<<<S_, 256, 0, stream>>>(xf, pos_emb);

  hipFuncSetAttribute((const void*)attention_kernel,
                      hipFuncAttributeMaxDynamicSharedMemorySize, ATTN_SMEM);

  for (int i = 0; i < DEPTH_; ++i) {
    layernorm_kernel<<<S_, 256, 0, stream>>>(xf, ln1_g + i * D_, ln1_b + i * D_, lnb);
    gemm_bf16_kernel<false, false, false, true>
        <<<dim3(3 * D_ / 256, S_ / 64), 256, 0, stream>>>(
            lnb, wqkv + (size_t)i * 3 * D_ * D_, qkv_b + i * 3 * D_, nullptr,
            nullptr, qkvb, S_, 3 * D_, D_);
    attention_kernel<<<dim3(32, NH_, 4), 128, ATTN_SMEM, stream>>>(qkvb, ob);
    gemm_bf16_kernel<false, true, true, false>
        <<<dim3(D_ / 256, S_ / 64), 256, 0, stream>>>(
            ob, wproj + (size_t)i * D_ * D_, proj_b + i * D_, xf, xf, nullptr,
            S_, D_, D_);
    layernorm_kernel<<<S_, 256, 0, stream>>>(xf, ln2_g + i * D_, ln2_b + i * D_, lnb);
    gemm_bf16_kernel<true, false, false, true>
        <<<dim3(FF_ / 256, S_ / 64), 256, 0, stream>>>(
            lnb, wfc1 + (size_t)i * FF_ * D_, fc1_b + i * FF_, nullptr, nullptr,
            hb, S_, FF_, D_);
    float* xout = (i == DEPTH_ - 1) ? (float*)d_out : xf;
    gemm_bf16_kernel<false, true, true, false>
        <<<dim3(D_ / 256, S_ / 64), 256, 0, stream>>>(
            hb, wfc2 + (size_t)i * D_ * FF_, fc2_b + i * D_, xf, xout, nullptr,
            S_, D_, FF_);
  }
}
